// RNN_71133248356918
// MI455X (gfx1250) — matrix operand order repaired and checked by reading, not compiled, not run on hardware
//
#include <hip/hip_runtime.h>

typedef _Float16 v16h __attribute__((ext_vector_type(16)));
typedef _Float16 v8h  __attribute__((ext_vector_type(8)));
typedef float    v8f  __attribute__((ext_vector_type(8)));

#define H     64
#define GATES 256        // 4*H
#define BTILE 32         // batch rows per workgroup (2 row-groups of 16)
#define HPAD  72         // padded row length (halves) for hA: 144B row stride

__device__ __forceinline__ float fast_sigmoid(float v) {
  return __builtin_amdgcn_rcpf(1.f + __expf(-v));
}
__device__ __forceinline__ float fast_tanh(float v) {
#if __has_builtin(__builtin_amdgcn_tanhf)
  return __builtin_amdgcn_tanhf(v);
#else
  return __builtin_fmaf(2.f, __builtin_amdgcn_rcpf(1.f + __expf(-2.f * v)), -1.f);
#endif
}

// ---- prep: W_hh f32 -> f16, fold biases ----------------------------------
__global__ void lstm_prep(const float* __restrict__ W_hh,
                          const float* __restrict__ b_ih,
                          const float* __restrict__ b_hh,
                          _Float16* __restrict__ Whh16,
                          float* __restrict__ bfold) {
  int i = blockIdx.x * blockDim.x + threadIdx.x;
  if (i < GATES * H) Whh16[i] = (_Float16)W_hh[i];
  if (i < GATES)     bfold[i] = b_ih[i] + b_hh[i];
}

// ---- main scan kernel -----------------------------------------------------
__launch_bounds__(256, 1)
__global__ void lstm_scan(const float* __restrict__ x,
                          const float* __restrict__ W_ih,
                          const _Float16* __restrict__ Whh16,
                          const float* __restrict__ bfold,
                          const float* __restrict__ W_out,
                          const float* __restrict__ b_out,
                          float* __restrict__ out,
                          int T) {
  __shared__ _Float16 hA[2][BTILE * HPAD];  // double-buffered h state, f16
  __shared__ float    xsh[2][2 * BTILE];    // double-buffered x_t
  __shared__ float    hfin[BTILE * H];      // final h, f32 (output head)

  const int tid  = threadIdx.x;
  const int lane = tid & 31;
  const int wv   = tid >> 5;       // 8 waves
  const int hi   = lane >> 4;      // half-wave select (0/1)
  const int lo   = lane & 15;
  const int rg   = wv >> 2;        // row group: rows rg*16 .. rg*16+15
  const int cg   = wv & 3;         // hidden-column group: cols cg*16 .. +15
  const int rowbase = blockIdx.x * BTILE;

  // ---- resident B fragments: ALL FOUR GATES for this wave's column slice --
  // Gate G occupies columns n = G*64 + cg*16 + lo.  B 32x16 f16, contiguous k order used here (A matches):
  // lane = column n; lanes 0-15 carry K=kc*32+0..15, lanes 16-31 carry
  // K=kc*32+16..31 -> 16 contiguous halves of W_hh row n (32B aligned load).
  v16h Bfrag[4][2];
  float bias[4], wih0[4], wih1[4];
  #pragma unroll
  for (int G = 0; G < 4; ++G) {
    int n = G * 64 + cg * 16 + lo;
    #pragma unroll
    for (int kc = 0; kc < 2; ++kc) {
      int k0 = kc * 32 + hi * 16;
      Bfrag[G][kc] = *(const v16h*)&Whh16[n * H + k0];
    }
    bias[G] = bfold[n];
    wih0[G] = W_ih[n * 2 + 0];
    wih1[G] = W_ih[n * 2 + 1];
  }

  // ---- state: c and h live in WMMA C-layout registers --------------------
  // lane element r: row m = rg*16 + r + 8*hi, column j = cg*16 + lo
  float creg[8] = {0.f,0.f,0.f,0.f,0.f,0.f,0.f,0.f};
  float hreg[8];
  for (int idx = tid; idx < BTILE * HPAD; idx += 256) hA[0][idx] = (_Float16)0.0f;
  // pre-stage x_0
  if (tid < 2 * BTILE) {
    int r = tid >> 1, i = tid & 1;
    xsh[0][tid] = x[((size_t)(rowbase + r) * T + 0) * 2 + i];
  }
  __syncthreads();

  for (int t = 0; t < T; ++t) {
    const int cb = t & 1, nb = cb ^ 1;

    // A 16x32 f16 fragments for this wave's row group (two 16B LDS loads per
    // K-chunk; lane row = rg*16+lo; lanes 0-15: K=c..c+15, lanes
    // 16-31: K=c+16..c+31)
    const int arow = (rg * 16 + lo) * HPAD;
    const int kl   = hi * 16;
    v8h a0lo = *(const v8h*)&hA[cb][arow + kl];
    v8h a0hi = *(const v8h*)&hA[cb][arow + kl + 8];
    v8h a1lo = *(const v8h*)&hA[cb][arow + 32 + kl];
    v8h a1hi = *(const v8h*)&hA[cb][arow + 32 + kl + 8];
    v16h A0 = __builtin_shufflevector(a0lo, a0hi, 0,1,2,3,4,5,6,7,8,9,10,11,12,13,14,15);
    v16h A1 = __builtin_shufflevector(a1lo, a1hi, 0,1,2,3,4,5,6,7,8,9,10,11,12,13,14,15);

    // gate pre-activations for all four gates of this column slice
    v8f acc[4];
    #pragma unroll
    for (int G = 0; G < 4; ++G) {
      #pragma unroll
      for (int r = 0; r < 8; ++r) {
        int m = rg * 16 + r + 8 * hi;
        acc[G][r] = bias[G] + xsh[cb][2*m] * wih0[G] + xsh[cb][2*m+1] * wih1[G];
      }
      acc[G] = __builtin_amdgcn_wmma_f32_16x16x32_f16(false, A0, false, Bfrag[G][0],
                                                      (short)0, acc[G], false, false);
      acc[G] = __builtin_amdgcn_wmma_f32_16x16x32_f16(false, A1, false, Bfrag[G][1],
                                                      (short)0, acc[G], false, false);
    }

    // stage x_{t+1} into the other buffer (no extra barrier needed)
    if (t + 1 < T && tid < 2 * BTILE) {
      int r = tid >> 1, i = tid & 1;
      xsh[nb][tid] = x[((size_t)(rowbase + r) * T + (t + 1)) * 2 + i];
      if (t + 2 < T)
        __builtin_prefetch(&x[((size_t)(rowbase + r) * T + (t + 2)) * 2 + i], 0, 0);
    }

    // fully in-register gate recombine: c = f*c + i*g ; h = o*tanh(c)
    #pragma unroll
    for (int r = 0; r < 8; ++r) {
      float iv = fast_sigmoid(acc[0][r]);
      float fv = fast_sigmoid(acc[1][r]);
      float gv = fast_tanh(acc[2][r]);
      float ov = fast_sigmoid(acc[3][r]);
      float c  = fv * creg[r] + iv * gv;
      creg[r]  = c;
      float hn = ov * fast_tanh(c);
      hreg[r]  = hn;
      hA[nb][(rg * 16 + r + 8 * hi) * HPAD + cg * 16 + lo] = (_Float16)hn;
    }
    __syncthreads();   // single barrier per step (buffers are double-buffered)
  }

  // ---- output head: out[b][o] = h_T . W_out[o] + b_out[o] ----------------
  #pragma unroll
  for (int r = 0; r < 8; ++r)
    hfin[(rg * 16 + r + 8 * hi) * H + cg * 16 + lo] = hreg[r];
  __syncthreads();

  if (tid < BTILE * 5) {
    int b = tid / 5, o = tid % 5;
    float s = b_out[o];
    #pragma unroll
    for (int j = 0; j < H; ++j) s += hfin[b * H + j] * W_out[o * H + j];
    out[(rowbase + b) * 5 + o] = s;
  }
}

extern "C" void kernel_launch(void* const* d_in, const int* in_sizes, int n_in,
                              void* d_out, int out_size, void* d_ws, size_t ws_size,
                              hipStream_t stream) {
  const float* x     = (const float*)d_in[0];
  const float* W_ih  = (const float*)d_in[1];
  const float* W_hh  = (const float*)d_in[2];
  const float* b_ih  = (const float*)d_in[3];
  const float* b_hh  = (const float*)d_in[4];
  const float* W_out = (const float*)d_in[5];
  const float* b_out = (const float*)d_in[6];
  float* out = (float*)d_out;

  const int B = out_size / 5;               // output is [1, B, 5]
  const int T = in_sizes[0] / (B * 2);      // x is [B, T, 2]

  _Float16* Whh16 = (_Float16*)d_ws;
  float*    bfold = (float*)((char*)d_ws + (size_t)GATES * H * sizeof(_Float16));

  lstm_prep<<<(GATES * H + 255) / 256, 256, 0, stream>>>(W_hh, b_ih, b_hh, Whh16, bfold);
  lstm_scan<<<B / BTILE, 256, 0, stream>>>(x, W_ih, Whh16, bfold, W_out, b_out, out, T);
}
